// GPTLanguageModel_46145128629002
// MI455X (gfx1250) — matrix operand order repaired and checked by reading, not compiled, not run on hardware
//
#include <hip/hip_runtime.h>
#include <math.h>

// ---------------- model constants ----------------
#define V_   50257
#define VP_  50304          // V padded to multiple of 64 (for unguarded tile loads)
#define D_   768
#define H_   12
#define HD_  64
#define L_   6
#define B_   4
#define T_   1024
#define BT_  4096
#define DFF_ 3072

// ---------------- WMMA / TDM types ----------------
typedef __attribute__((ext_vector_type(16))) __bf16 v16bf;
typedef __attribute__((ext_vector_type(8)))  float  v8f;
typedef __attribute__((ext_vector_type(4)))  unsigned int v4u;
typedef __attribute__((ext_vector_type(8)))  int    v8i;
typedef __attribute__((ext_vector_type(4)))  int    v4i;

union FragU { v16bf v; unsigned int u[8]; };

#if defined(__has_builtin)
#  if __has_builtin(__builtin_amdgcn_tensor_load_to_lds) && __has_builtin(__builtin_amdgcn_s_wait_tensorcnt)
#    define HAVE_TDM 1
#  else
#    define HAVE_TDM 0
#  endif
#else
#  define HAVE_TDM 0
#endif

__device__ __forceinline__ unsigned short f2bf(float f) {
    unsigned int u = __float_as_uint(f);
    unsigned int r = u + 0x7FFFu + ((u >> 16) & 1u);   // round-to-nearest-even
    return (unsigned short)(r >> 16);
}

// LDS byte offset of a __shared__ object: flat LDS-aperture addresses carry the
// LDS offset in addr[31:0] (ISA 10.2 per-aperture mapping).
__device__ __forceinline__ unsigned lds_off(const void* p) {
    return (unsigned)(unsigned long long)p;
}

// Async DMA: 16 bytes global -> LDS per enabled lane (VGLOBAL op 98, GV mode).
__device__ __forceinline__ void async_copy16(void* lds_dst, const void* gsrc) {
    asm volatile("global_load_async_to_lds_b128 %0, %1, off"
                 :: "v"(lds_off(lds_dst)), "v"(gsrc)
                 : "memory");
}
__device__ __forceinline__ void wait_async() {
    asm volatile("s_wait_asynccnt 0x0" ::: "memory");
}

// TDM: load a 2D tile (tile_k x tile_n rows) of 2-byte elements from a row-major
// tensor (row length/stride = k elements) into contiguous LDS.  D# layout per
// ISA 8.3 (group0: count/lds_addr/global_addr/type) and 8.4 (group1: data_size,
// tensor dims, tile dims, strides).  Issued by one wave; tracked by TENSORcnt.
__device__ __forceinline__ void tdm_load_tile_2d(
    void* lds_dst, const void* gsrc, unsigned k, unsigned nrows,
    unsigned tile_k, unsigned tile_n)
{
#if HAVE_TDM
    const unsigned long long ga = (unsigned long long)gsrc;
    v4u g0;
    g0[0] = 1u;                                              // count=1 (valid user D#)
    g0[1] = lds_off(lds_dst);                                // lds_addr
    g0[2] = (unsigned)ga;                                    // global_addr[31:0]
    g0[3] = (unsigned)((ga >> 32) & 0x1FFFFFFull) | (2u << 30); // addr[56:32] | type=2
    v8i g1;
    g1[0] = (int)(1u << 16);                                 // data_size=1 -> 2 bytes
    g1[1] = (int)((k & 0xFFFFu) << 16);                      // tensor_dim0[15:0] @ bit48
    g1[2] = (int)(((k >> 16) & 0xFFFFu) | ((nrows & 0xFFFFu) << 16)); // dim0 hi | dim1 lo
    g1[3] = (int)(((nrows >> 16) & 0xFFFFu) | (tile_k << 16));        // dim1 hi | tile_dim0
    g1[4] = (int)(tile_n & 0xFFFFu);                         // tile_dim1 (tile_dim2=0)
    g1[5] = (int)k;                                          // tensor_dim0_stride[31:0]
    g1[6] = 0;                                               // stride0 hi | stride1 lo
    g1[7] = 0;
    const v4i z4 = {0, 0, 0, 0};
#  if __clang_major__ >= 23
    const v8i z8 = {0, 0, 0, 0, 0, 0, 0, 0};
    __builtin_amdgcn_tensor_load_to_lds(g0, g1, z4, z4, z8, 0);
#  else
    __builtin_amdgcn_tensor_load_to_lds(g0, g1, z4, z4, 0);
#  endif
#else
    (void)lds_dst; (void)gsrc; (void)k; (void)nrows; (void)tile_k; (void)tile_n;
#endif
}

// A-matrix fragment (16x32 bf16, ISA 7.12.2): lane lm holds row M=lm.
__device__ __forceinline__ v16bf frag_a(const unsigned short* row, int lh) {
    const unsigned int* r32 = (const unsigned int*)row;
    FragU f;
    const int base = lh * 8;
#pragma unroll
    for (int j = 0; j < 4; ++j) f.u[j]     = r32[base + j];
#pragma unroll
    for (int j = 0; j < 4; ++j) f.u[4 + j] = r32[base + 4 + j];
    return f.v;
}

// B-matrix fragment (32x16 bf16): lane lm = column N; half selects K 0..15 / 16..31.
__device__ __forceinline__ v16bf frag_b(const unsigned short* col, int lh) {
    const unsigned int* r32 = (const unsigned int*)col;
    FragU f;
#pragma unroll
    for (int j = 0; j < 8; ++j) f.u[j] = r32[lh * 8 + j];
    return f.v;
}

__device__ __forceinline__ v8f wmma_bf16(v16bf a, v16bf b, v8f c) {
    return __builtin_amdgcn_wmma_f32_16x16x32_bf16(
        false, a, false, b, (short)0, c, false, false);
}

// ---------------- weight convert + transpose: W[K][N] f32 -> Wt[N][K] bf16 ----
__global__ __launch_bounds__(256) void convert_transpose_kernel(
    const float* __restrict__ W, unsigned short* __restrict__ Wt,
    int K, int N, int Npad)
{
    __shared__ float tile[32][33];
    const int n0 = blockIdx.x * 32;
    const int k0 = blockIdx.y * 32;
    const int tx = threadIdx.x & 31, ty = threadIdx.x >> 5;
#pragma unroll
    for (int i = 0; i < 4; ++i) {
        const int kk = ty + i * 8;
        const int n  = n0 + tx;
        tile[kk][tx] = (n < N) ? W[(size_t)(k0 + kk) * N + n] : 0.0f;
    }
    __syncthreads();
#pragma unroll
    for (int i = 0; i < 4; ++i) {
        const int nn = ty + i * 8;
        if (n0 + nn < Npad)
            Wt[(size_t)(n0 + nn) * K + k0 + tx] = f2bf(tile[tx][nn]);
    }
}

// ---------------- embedding ----------------
__global__ __launch_bounds__(256) void embed_kernel(
    const float* __restrict__ tok, const float* __restrict__ pos,
    const int* __restrict__ idx, float* __restrict__ x)
{
    const int row = blockIdx.x;
    const int t   = row % T_;
    const int id  = idx[row];
    for (int d = threadIdx.x; d < D_; d += 256)
        x[(size_t)row * D_ + d] = tok[(size_t)id * D_ + d] + pos[(size_t)t * D_ + d];
}

// ---------------- layernorm: f32 in, bf16 out ----------------
__global__ __launch_bounds__(256) void layernorm_kernel(
    const float* __restrict__ in, const float* __restrict__ w,
    const float* __restrict__ b, unsigned short* __restrict__ out)
{
    __shared__ float red[256];
    const int row = blockIdx.x, tid = threadIdx.x;
    const float* xr = in + (size_t)row * D_;

    float s = 0.f, ss = 0.f;
    for (int d = tid; d < D_; d += 256) { float v = xr[d]; s += v; ss += v * v; }

    red[tid] = s; __syncthreads();
    for (int off = 128; off > 0; off >>= 1) {
        if (tid < off) red[tid] += red[tid + off];
        __syncthreads();
    }
    const float mean = red[0] * (1.0f / D_);
    __syncthreads();

    red[tid] = ss; __syncthreads();
    for (int off = 128; off > 0; off >>= 1) {
        if (tid < off) red[tid] += red[tid + off];
        __syncthreads();
    }
    const float var  = red[0] * (1.0f / D_) - mean * mean;
    const float rstd = rsqrtf(var + 1e-5f);

    for (int d = tid; d < D_; d += 256)
        out[(size_t)row * D_ + d] = f2bf((xr[d] - mean) * rstd * w[d] + b[d]);
}

// ---------------- tiled bf16 WMMA GEMM, double-buffered DMA pipeline ---------
// C[M,N] = A[M,K] @ Wt[N,K]^T (+epilogue). A tiles via per-lane async DMA,
// B (weight) tiles via one TDM tensor_load_to_lds per K-step (wave 0).
// EPI: 0 none, 1 +bias, 2 +bias+GELU, 3 +bias+residual(fp32 R)
// OUTBF: 0 -> fp32 C, 1 -> bf16 C
template <int EPI, int OUTBF>
__global__ __launch_bounds__(128) void gemm_bf16_kernel(
    const unsigned short* __restrict__ A,   // [M][K] bf16
    const unsigned short* __restrict__ Wt,  // [Nrows][K] bf16 (pre-transposed)
    const float* __restrict__ bias, const float* __restrict__ R,
    void* __restrict__ Cout, int N, int K, int Nrows)
{
    __shared__ __attribute__((aligned(16))) unsigned short As[2][64][32];
    __shared__ __attribute__((aligned(16))) unsigned short Bs[2][64][32];

    const int tid  = threadIdx.x;
    const int wave = tid >> 5;
    const int lane = tid & 31;
    const int lh   = lane >> 4;
    const int lm   = lane & 15;
    const int rowBase = blockIdx.y * 64;
    const int colBase = blockIdx.x * 64;

    auto issue_tile = [&](int buf, int k0) {
#pragma unroll
        for (int i = 0; i < 2; ++i) {                       // A: 256 x 16B chunks
            const int ch = tid + i * 128;
            const int r = ch >> 2, ko = (ch & 3) * 8;
            async_copy16(&As[buf][r][ko], A + (size_t)(rowBase + r) * K + k0 + ko);
        }
#if HAVE_TDM
        if (wave == 0)                                      // B: one TDM 2D tile
            tdm_load_tile_2d(&Bs[buf][0][0], Wt + (size_t)colBase * K + k0,
                             (unsigned)K, (unsigned)Nrows, 32u, 64u);
#else
#pragma unroll
        for (int i = 0; i < 2; ++i) {
            const int ch = tid + i * 128;
            const int n = ch >> 2, ko = (ch & 3) * 8;
            async_copy16(&Bs[buf][n][ko], Wt + (size_t)(colBase + n) * K + k0 + ko);
        }
#endif
    };
    auto wait_tiles = [&]() {
        wait_async();
#if HAVE_TDM
        if (wave == 0) __builtin_amdgcn_s_wait_tensorcnt(0);
#endif
    };

    v8f acc[4];
#pragma unroll
    for (int nt = 0; nt < 4; ++nt)
#pragma unroll
        for (int i = 0; i < 8; ++i) acc[nt][i] = 0.0f;

    issue_tile(0, 0);
    wait_tiles();
    __syncthreads();

    for (int k0 = 0; k0 < K; k0 += 32) {
        const int  cur  = (k0 >> 5) & 1;
        const bool more = (k0 + 32) < K;
        if (more) issue_tile(cur ^ 1, k0 + 32);             // overlap DMA with WMMA
        if (k0 + 64 < K)                                    // global_prefetch_b8
            __builtin_prefetch(A + (size_t)(rowBase + (tid & 63)) * K + k0 + 64);

        v16bf a = frag_a(&As[cur][16 * wave + lm][0], lh);
#pragma unroll
        for (int nt = 0; nt < 4; ++nt) {
            v16bf b = frag_b(&Bs[cur][16 * nt + lm][0], lh);
            acc[nt] = wmma_bf16(a, b, acc[nt]);
        }

        if (more) wait_tiles();
        __syncthreads();   // next tile visible to all; closes WAR on buffer reuse
    }

#pragma unroll
    for (int nt = 0; nt < 4; ++nt) {
        const int col = colBase + 16 * nt + lm;
        if (col < N) {
            const float bv = (EPI >= 1) ? bias[col] : 0.0f;
#pragma unroll
            for (int r = 0; r < 8; ++r) {
                const int row = rowBase + 16 * wave + r + 8 * lh;
                float val = acc[nt][r] + bv;
                if (EPI == 2) val = 0.5f * val * (1.0f + erff(val * 0.70710678118654752f));
                if (EPI == 3) val += R[(size_t)row * N + col];
                if (OUTBF)
                    ((unsigned short*)Cout)[(size_t)row * N + col] = f2bf(val);
                else
                    ((float*)Cout)[(size_t)row * N + col] = val;
            }
        }
    }
}

// ---------------- flash attention (bf16 in/out), one wave per (b,h,q-tile) ---
__global__ __launch_bounds__(32) void attn_kernel(
    const unsigned short* __restrict__ q, const unsigned short* __restrict__ k,
    const unsigned short* __restrict__ v, unsigned short* __restrict__ o)
{
    __shared__ __attribute__((aligned(16))) unsigned short Qs[16][64];
    __shared__ __attribute__((aligned(16))) unsigned short Ks[32][64];
    __shared__ __attribute__((aligned(16))) unsigned short Vs[64][32];
    __shared__ __attribute__((aligned(16))) unsigned short Ps[16][32];

    const int lane = threadIdx.x;
    const int lh = lane >> 4, lm = lane & 15;
    const int qt = blockIdx.x;
    const int bh = blockIdx.y;
    const int bi = bh / H_, hi = bh % H_;
    const size_t base = (size_t)bi * T_ * D_ + (size_t)hi * HD_;

    for (int ch = lane; ch < 128; ch += 32) {
        const int r = ch >> 3, off = (ch & 7) * 8;
        async_copy16(&Qs[r][off], q + base + (size_t)(16 * qt + r) * D_ + off);
    }

    float m[8], lsum[8];
    v8f oacc[4];
#pragma unroll
    for (int r = 0; r < 8; ++r) { m[r] = -3.0e38f; lsum[r] = 0.0f; }
#pragma unroll
    for (int nt = 0; nt < 4; ++nt)
#pragma unroll
        for (int i = 0; i < 8; ++i) oacc[nt][i] = 0.0f;

    const int jmax = (16 * qt + 15) / 32;
    for (int j = 0; j <= jmax; ++j) {
        for (int ch = lane; ch < 256; ch += 32) {
            const int r = ch >> 3, off = (ch & 7) * 8;
            async_copy16(&Ks[r][off], k + base + (size_t)(32 * j + r) * D_ + off);
        }
        for (int i = lane; i < 2048; i += 32) {
            const int d = i >> 5, r = i & 31;
            Vs[d][r] = v[base + (size_t)(32 * j + r) * D_ + d];
        }
        wait_async();
        __syncthreads();

        v8f s0, s1;
#pragma unroll
        for (int i = 0; i < 8; ++i) { s0[i] = 0.f; s1[i] = 0.f; }
#pragma unroll
        for (int db = 0; db < 64; db += 32) {
            v16bf qa  = frag_a(&Qs[lm][db], lh);
            v16bf kf0 = frag_b(&Ks[lm][db], lh);
            v16bf kf1 = frag_b(&Ks[16 + lm][db], lh);
            s0 = wmma_bf16(qa, kf0, s0);
            s1 = wmma_bf16(qa, kf1, s1);
        }

        const int rowg0 = 16 * qt + 8 * lh;
        const int col0 = 32 * j + lm, col1 = col0 + 16;
        float p0[8], p1[8], alpha[8];
#pragma unroll
        for (int r = 0; r < 8; ++r) {
            const int rowg = rowg0 + r;
            float a0 = (col0 <= rowg) ? s0[r] * 0.125f : -3.0e38f;
            float a1 = (col1 <= rowg) ? s1[r] * 0.125f : -3.0e38f;
            float mx = fmaxf(a0, a1);
            mx = fmaxf(mx, __shfl_xor(mx, 1));
            mx = fmaxf(mx, __shfl_xor(mx, 2));
            mx = fmaxf(mx, __shfl_xor(mx, 4));
            mx = fmaxf(mx, __shfl_xor(mx, 8));
            const float mn = fmaxf(m[r], mx);
            alpha[r] = expf(m[r] - mn);
            p0[r] = expf(a0 - mn);
            p1[r] = expf(a1 - mn);
            float rs = p0[r] + p1[r];
            rs += __shfl_xor(rs, 1);
            rs += __shfl_xor(rs, 2);
            rs += __shfl_xor(rs, 4);
            rs += __shfl_xor(rs, 8);
            lsum[r] = lsum[r] * alpha[r] + rs;
            m[r] = mn;
        }

#pragma unroll
        for (int r = 0; r < 8; ++r) {
            Ps[r + 8 * lh][lm]      = f2bf(p0[r]);
            Ps[r + 8 * lh][16 + lm] = f2bf(p1[r]);
        }
        __syncthreads();

#pragma unroll
        for (int nt = 0; nt < 4; ++nt)
#pragma unroll
            for (int r = 0; r < 8; ++r) oacc[nt][r] *= alpha[r];

        v16bf pa = frag_a(&Ps[lm][0], lh);
#pragma unroll
        for (int nt = 0; nt < 4; ++nt) {
            v16bf vb = frag_b(&Vs[16 * nt + lm][0], lh);
            oacc[nt] = wmma_bf16(pa, vb, oacc[nt]);
        }
        __syncthreads();
    }

#pragma unroll
    for (int nt = 0; nt < 4; ++nt) {
        const int d = 16 * nt + lm;
#pragma unroll
        for (int r = 0; r < 8; ++r) {
            const int rowg = 16 * qt + r + 8 * lh;
            o[base + (size_t)rowg * D_ + d] = f2bf(oacc[nt][r] / lsum[r]);
        }
    }
}

// ---------------- loss: log-softmax + NLL mean ----------------
__global__ void zero_kernel(float* p) { if (threadIdx.x == 0 && blockIdx.x == 0) *p = 0.0f; }

__global__ __launch_bounds__(256) void loss_kernel(
    const float* __restrict__ logits, const int* __restrict__ targets,
    float* __restrict__ loss)
{
    __shared__ float red[256];
    const int row = blockIdx.x, tid = threadIdx.x;
    const float* lr = logits + (size_t)row * V_;

    float mx = -3.0e38f;
    for (int i = tid; i < V_; i += 256) mx = fmaxf(mx, lr[i]);
    red[tid] = mx; __syncthreads();
    for (int off = 128; off > 0; off >>= 1) {
        if (tid < off) red[tid] = fmaxf(red[tid], red[tid + off]);
        __syncthreads();
    }
    mx = red[0]; __syncthreads();

    float s = 0.f;
    for (int i = tid; i < V_; i += 256) s += expf(lr[i] - mx);
    red[tid] = s; __syncthreads();
    for (int off = 128; off > 0; off >>= 1) {
        if (tid < off) red[tid] += red[tid + off];
        __syncthreads();
    }

    if (tid == 0) {
        const float lse = mx + logf(red[0]);
        const float nll = lse - lr[targets[row]];
        atomicAdd(loss, nll * (1.0f / (float)BT_));
    }
}

// ---------------- launch ----------------
extern "C" void kernel_launch(void* const* d_in, const int* in_sizes, int n_in,
                              void* d_out, int out_size, void* d_ws, size_t ws_size,
                              hipStream_t stream) {
    (void)in_sizes; (void)n_in; (void)out_size; (void)ws_size;

    const float* tok_emb = (const float*)d_in[0];
    const float* pos_emb = (const float*)d_in[1];
    const float* ln1_w   = (const float*)d_in[2];
    const float* ln1_b   = (const float*)d_in[3];
    const float* ln2_w   = (const float*)d_in[4];
    const float* ln2_b   = (const float*)d_in[5];
    const float* wq      = (const float*)d_in[6];
    const float* wk      = (const float*)d_in[7];
    const float* wv      = (const float*)d_in[8];
    const float* wproj   = (const float*)d_in[9];
    const float* bproj   = (const float*)d_in[10];
    const float* w1      = (const float*)d_in[11];
    const float* b1      = (const float*)d_in[12];
    const float* w2      = (const float*)d_in[13];
    const float* b2      = (const float*)d_in[14];
    const float* lnf_w   = (const float*)d_in[15];
    const float* lnf_b   = (const float*)d_in[16];
    const float* lm_w    = (const float*)d_in[17];
    const float* lm_b    = (const float*)d_in[18];
    const int*   index   = (const int*)d_in[19];
    const int*   targets = (const int*)d_in[20];

    float* out    = (float*)d_out;
    float* logits = out;
    float* loss   = out + (size_t)BT_ * V_;

    // ---- workspace carve-up ----
    const size_t BTD = (size_t)BT_ * D_;
    const size_t SDD = (size_t)D_ * D_;
    const size_t SDF = (size_t)D_ * DFF_;
    float* ws = (float*)d_ws;
    float* x  = ws;                                    // [BT][D] f32 residual
    unsigned short* us = (unsigned short*)(x + BTD);
    unsigned short* hb = us;              us += BTD;
    unsigned short* qb = us;              us += BTD;
    unsigned short* kb = us;              us += BTD;
    unsigned short* vb = us;              us += BTD;
    unsigned short* ab = us;              us += BTD;
    unsigned short* fb = us;              us += (size_t)BT_ * DFF_;
    unsigned short* wqT = us;             us += (size_t)L_ * SDD;
    unsigned short* wkT = us;             us += (size_t)L_ * SDD;
    unsigned short* wvT = us;             us += (size_t)L_ * SDD;
    unsigned short* wpT = us;             us += (size_t)L_ * SDD;
    unsigned short* w1T = us;             us += (size_t)L_ * SDF;  // [DFF][D]
    unsigned short* w2T = us;             us += (size_t)L_ * SDF;  // [D][DFF]
    unsigned short* lmT = us;             us += (size_t)VP_ * D_;  // [VP][D]

    // ---- weight convert + transpose (once per call) ----
    const dim3 ctDD(D_ / 32, D_ / 32);
    const dim3 ctDF(DFF_ / 32, D_ / 32);
    const dim3 ctFD(D_ / 32, DFF_ / 32);
    const dim3 ctDV(VP_ / 32, D_ / 32);
    for (int l = 0; l < L_; ++l) {
        convert_transpose_kernel<<<ctDD, 256, 0, stream>>>(wq + l * SDD,    wqT + l * SDD, D_, D_, D_);
        convert_transpose_kernel<<<ctDD, 256, 0, stream>>>(wk + l * SDD,    wkT + l * SDD, D_, D_, D_);
        convert_transpose_kernel<<<ctDD, 256, 0, stream>>>(wv + l * SDD,    wvT + l * SDD, D_, D_, D_);
        convert_transpose_kernel<<<ctDD, 256, 0, stream>>>(wproj + l * SDD, wpT + l * SDD, D_, D_, D_);
        convert_transpose_kernel<<<ctDF, 256, 0, stream>>>(w1 + l * SDF,    w1T + l * SDF, D_, DFF_, DFF_);
        convert_transpose_kernel<<<ctFD, 256, 0, stream>>>(w2 + l * SDF,    w2T + l * SDF, DFF_, D_, D_);
    }
    convert_transpose_kernel<<<ctDV, 256, 0, stream>>>(lm_w, lmT, D_, V_, VP_);

    // ---- forward ----
    const dim3 gemmBlk(128);
    const dim3 gDD(D_ / 64,   BT_ / 64);
    const dim3 gDF(DFF_ / 64, BT_ / 64);
    const dim3 gDV((V_ + 63) / 64, BT_ / 64);

    embed_kernel<<<BT_, 256, 0, stream>>>(tok_emb, pos_emb, index, x);

    for (int l = 0; l < L_; ++l) {
        const size_t oD = (size_t)l * D_;

        layernorm_kernel<<<BT_, 256, 0, stream>>>(x, ln1_w + oD, ln1_b + oD, hb);

        gemm_bf16_kernel<0, 1><<<gDD, gemmBlk, 0, stream>>>(hb, wqT + l * SDD, nullptr, nullptr, qb, D_, D_, D_);
        gemm_bf16_kernel<0, 1><<<gDD, gemmBlk, 0, stream>>>(hb, wkT + l * SDD, nullptr, nullptr, kb, D_, D_, D_);
        gemm_bf16_kernel<0, 1><<<gDD, gemmBlk, 0, stream>>>(hb, wvT + l * SDD, nullptr, nullptr, vb, D_, D_, D_);

        attn_kernel<<<dim3(T_ / 16, B_ * H_), 32, 0, stream>>>(qb, kb, vb, ab);

        gemm_bf16_kernel<3, 0><<<gDD, gemmBlk, 0, stream>>>(ab, wpT + l * SDD, bproj + oD, x, x, D_, D_, D_);

        layernorm_kernel<<<BT_, 256, 0, stream>>>(x, ln2_w + oD, ln2_b + oD, hb);

        gemm_bf16_kernel<2, 1><<<gDF, gemmBlk, 0, stream>>>(hb, w1T + l * SDF, b1 + (size_t)l * DFF_,
                                                            nullptr, fb, DFF_, D_, DFF_);
        gemm_bf16_kernel<3, 0><<<gDD, gemmBlk, 0, stream>>>(fb, w2T + l * SDF, b2 + oD, x, x, D_, DFF_, D_);
    }

    layernorm_kernel<<<BT_, 256, 0, stream>>>(x, lnf_w, lnf_b, hb);

    gemm_bf16_kernel<1, 0><<<gDV, gemmBlk, 0, stream>>>(hb, lmT, lm_b, nullptr, logits, V_, D_, VP_);

    zero_kernel<<<1, 1, 0, stream>>>(loss);
    loss_kernel<<<BT_, 256, 0, stream>>>(logits, targets, loss);
}
